// FastAttention_2637109920371
// MI455X (gfx1250) — matrix operand order repaired and checked by reading, not compiled, not run on hardware
//
#include <hip/hip_runtime.h>

#define DEVFN __device__ __forceinline__

typedef __bf16 bf16;
typedef __attribute__((ext_vector_type(8)))  __bf16 v8bf;
typedef __attribute__((ext_vector_type(16))) __bf16 v16bf;
typedef __attribute__((ext_vector_type(8)))  float  v8f;

static constexpr int Bn = 2;
static constexpr int Sn = 16384;
static constexpr int En = 512;
static constexpr int Hn = 8;
static constexpr int Dn = 64;   // head dim
static constexpr int Mn = 64;   // random features
static constexpr int NR = Bn * Sn;        // 32768 rows
static constexpr int DP = 80;             // padded KV cols: 64 V + 1 ones + 15 pad
static constexpr float CLAMP_MAX = 50.0f;
static constexpr float EPSV = 1e-8f;

// ---------------- WMMA fragment helpers (CDNA5 16x16x32 bf16) ----------------
// A (16x32, MxK): lanes 0-15 row M=lane, elems 0..7 = K k0..k0+7, 8..15 = K k0+16..k0+23
//                 lanes 16-31 row M=lane-16, K halves shifted by 8.
// B from BT[N,K]: lanes 0-15 col N=lane, elems 0..7 = K k0..k0+7, 8..15 = K k0+16..k0+23; lanes 16-31 col lane-16, K halves shifted by 8.
// C/D f32: VGPR r -> row r (lanes 0-15) / row r+8 (lanes 16-31), col = lane&15.

DEVFN v16bf concat8(v8bf lo, v8bf hi) {
  v16bf f;
#pragma unroll
  for (int i = 0; i < 8; ++i) { f[i] = lo[i]; f[i + 8] = hi[i]; }
  return f;
}

DEVFN v16bf load_frag_a(const bf16* __restrict__ A, int lda, int k0, int lane) {
  const int half = (lane >> 4) & 1;
  const int r    = lane & 15;
  const bf16* p  = A + (size_t)r * lda + (k0 + half * 8);
  v8bf lo = *(const v8bf*)(p);        // K = k0+half*8 .. +7
  v8bf hi = *(const v8bf*)(p + 16);   // K = k0+16+half*8 .. +7
  return concat8(lo, hi);
}

DEVFN v16bf load_frag_b(const bf16* __restrict__ BT, int ldb, int k0, int lane) {
  const int half = (lane >> 4) & 1;
  const int n    = lane & 15;
  const bf16* p  = BT + (size_t)n * ldb + (k0 + half * 8);
  v8bf lo = *(const v8bf*)(p);
  v8bf hi = *(const v8bf*)(p + 16);
  return concat8(lo, hi);
}

DEVFN v8f wmma_bf16(v16bf a, v16bf b, v8f c) {
  return __builtin_amdgcn_wmma_f32_16x16x32_bf16(false, a, false, b, (short)0, c,
                                                 false, false);
}

// ---------------- utility kernels ----------------
__global__ void k_cvt_bf16(const float* __restrict__ src, bf16* __restrict__ dst, int n) {
  int i = blockIdx.x * 256 + threadIdx.x;
  if (i < n) dst[i] = (bf16)src[i];
}

// dst[C][R] = src[R][C]   (bf16 transpose-convert)
__global__ void k_transpose_bf16(const float* __restrict__ src, bf16* __restrict__ dst,
                                 int R, int C) {
  int i = blockIdx.x * 256 + threadIdx.x;
  if (i >= R * C) return;
  int r = i / C, c = i % C;
  dst[(size_t)c * R + r] = (bf16)src[i];
}

__global__ void k_zero_f32(float* __restrict__ p, int n) {
  int i = blockIdx.x * 256 + threadIdx.x;
  if (i < n) p[i] = 0.0f;
}

// ---------------- 1) fused Q/K/V projection GEMM ----------------
// Y[NR,En] = xb @ W^T-stored-as-BT[N,K]; z selects Q/K/V. Q,K scaled 1/sqrt(D)=0.125.
// V written transposed: VT[((b*H+h)*D + d)*S + s]
__global__ __launch_bounds__(256) void k_proj(
    const bf16* __restrict__ xb,
    const bf16* __restrict__ WTq, const bf16* __restrict__ WTk, const bf16* __restrict__ WTv,
    bf16* __restrict__ Qb, bf16* __restrict__ Kb, bf16* __restrict__ VT) {
  const int w = threadIdx.x >> 5, lane = threadIdx.x & 31;
  const int row0 = blockIdx.x * 128 + (w >> 1) * 32;
  const int col0 = blockIdx.y * 64 + (w & 1) * 32;
  const int which = blockIdx.z;
  const bf16* BT = (which == 0) ? WTq : (which == 1) ? WTk : WTv;

  const bf16* A = xb + (size_t)row0 * En;
  v8f acc[2][2] = {};
#pragma unroll 4
  for (int k0 = 0; k0 < En; k0 += 32) {
    v16bf a0 = load_frag_a(A, En, k0, lane);
    v16bf a1 = load_frag_a(A + (size_t)16 * En, En, k0, lane);
    v16bf b0 = load_frag_b(BT + (size_t)col0 * En, En, k0, lane);
    v16bf b1 = load_frag_b(BT + (size_t)(col0 + 16) * En, En, k0, lane);
    acc[0][0] = wmma_bf16(a0, b0, acc[0][0]);
    acc[0][1] = wmma_bf16(a0, b1, acc[0][1]);
    acc[1][0] = wmma_bf16(a1, b0, acc[1][0]);
    acc[1][1] = wmma_bf16(a1, b1, acc[1][1]);
  }
  const float scale = (which == 2) ? 1.0f : 0.125f;
#pragma unroll
  for (int i = 0; i < 2; ++i)
#pragma unroll
    for (int j = 0; j < 2; ++j)
#pragma unroll
      for (int r = 0; r < 8; ++r) {
        int row = row0 + i * 16 + r + ((lane >> 4) & 1) * 8;
        int col = col0 + j * 16 + (lane & 15);
        float v = acc[i][j][r] * scale;
        if (which == 0) {
          Qb[(size_t)row * En + col] = (bf16)v;
        } else if (which == 1) {
          Kb[(size_t)row * En + col] = (bf16)v;
        } else {
          int b = row >> 14, s = row & (Sn - 1);
          int h = col >> 6,  d = col & 63;
          VT[(((size_t)(b * Hn + h)) * Dn + d) * Sn + s] = (bf16)v;
        }
      }
}

// ---------------- 2) per-(row,head) squared L2 norms ----------------
__global__ void k_norm2(const bf16* __restrict__ Qb, const bf16* __restrict__ Kb,
                        float* __restrict__ qn2, float* __restrict__ kn2) {
  int idx = blockIdx.x * 256 + threadIdx.x;   // row*H + h
  if (idx >= NR * Hn) return;
  const bf16* src = blockIdx.z ? Kb : Qb;
  float* dst = blockIdx.z ? kn2 : qn2;
  int row = idx >> 3, h = idx & 7;
  const bf16* p = src + (size_t)row * En + h * Dn;
  float sum = 0.0f;
#pragma unroll
  for (int i = 0; i < 8; ++i) {
    v8bf v = *(const v8bf*)(p + i * 8);
#pragma unroll
    for (int t = 0; t < 8; ++t) { float x = (float)v[t]; sum += x * x; }
  }
  dst[idx] = sum;
}

// ---------------- 3) positive random feature map (GEMM K=64 + epilogue) -------
// z in [0,32): z<16 -> Q path (out row-major [s,m]); z>=16 -> K path (out [m,s]).
__global__ __launch_bounds__(256) void k_phi(
    const bf16* __restrict__ Qb, const bf16* __restrict__ Kb,
    const bf16* __restrict__ omqT, const bf16* __restrict__ omkT,
    const float* __restrict__ qn2, const float* __restrict__ kn2,
    bf16* __restrict__ Qphi, bf16* __restrict__ KphiT) {
  const int w = threadIdx.x >> 5, lane = threadIdx.x & 31;
  const int z = blockIdx.z;
  const int is_k = (z >= Bn * Hn);
  const int bh = z & (Bn * Hn - 1);
  const int b = bh >> 3, h = bh & 7;
  const int s0 = blockIdx.x * 128 + (w >> 1) * 32;
  const int m0 = (w & 1) * 32;

  const bf16* A  = (is_k ? Kb : Qb) + ((size_t)b * Sn + s0) * En + h * Dn;
  const bf16* BT = is_k ? omkT : omqT;          // [m][d], ldb = 64
  const float* n2 = is_k ? kn2 : qn2;

  v8f acc[2][2] = {};
#pragma unroll
  for (int k0 = 0; k0 < Dn; k0 += 32) {
    v16bf a0 = load_frag_a(A, En, k0, lane);
    v16bf a1 = load_frag_a(A + (size_t)16 * En, En, k0, lane);
    v16bf b0 = load_frag_b(BT + (size_t)m0 * Dn, Dn, k0, lane);
    v16bf b1 = load_frag_b(BT + (size_t)(m0 + 16) * Dn, Dn, k0, lane);
    acc[0][0] = wmma_bf16(a0, b0, acc[0][0]);
    acc[0][1] = wmma_bf16(a0, b1, acc[0][1]);
    acc[1][0] = wmma_bf16(a1, b0, acc[1][0]);
    acc[1][1] = wmma_bf16(a1, b1, acc[1][1]);
  }
#pragma unroll
  for (int i = 0; i < 2; ++i)
#pragma unroll
    for (int j = 0; j < 2; ++j)
#pragma unroll
      for (int r = 0; r < 8; ++r) {
        int s = s0 + i * 16 + r + ((lane >> 4) & 1) * 8;
        int m = m0 + j * 16 + (lane & 15);
        float nn = n2[(size_t)(b * Sn + s) * Hn + h];
        float f = fminf(acc[i][j][r], CLAMP_MAX);
        float phi = fmaxf(expf(-0.5f * nn), EPSV) * expf(f);
        if (is_k) KphiT[((size_t)bh * Mn + m) * Sn + s] = (bf16)phi;
        else      Qphi[((size_t)bh * Sn + s) * Mn + m] = (bf16)phi;
      }
}

// ---------------- 4) KV = Kphi^T V (+ ones column for K_ones) -----------------
// C[m, n] per (b,h): n<64 -> sum_s Kphi[s,m] V[s,n];  n==64 -> sum_s Kphi[s,m]
__global__ __launch_bounds__(256) void k_kv(
    const bf16* __restrict__ KphiT, const bf16* __restrict__ VT,
    float* __restrict__ KVp) {
  const int w = threadIdx.x >> 5, lane = threadIdx.x & 31;
  const int bh = blockIdx.z;
  const int m0 = (w & 1) * 32;
  const int kbase = blockIdx.x * 2048 + (w >> 1) * 512;

  const bf16* A  = KphiT + ((size_t)bh * Mn + m0) * Sn;  // lda = S
  const bf16* Bv = VT + (size_t)bh * Dn * Sn;            // BT[n][s], ldb = S

  v16bf onesf;
  {
    bf16 v = (bf16)(((lane & 15) == 0) ? 1.0f : 0.0f);
#pragma unroll
    for (int i = 0; i < 16; ++i) onesf[i] = v;
  }

  v8f acc[2][5] = {};
  for (int it = 0; it < 16; ++it) {
    int k0 = kbase + it * 32;
    v16bf a0 = load_frag_a(A, Sn, k0, lane);
    v16bf a1 = load_frag_a(A + (size_t)16 * Sn, Sn, k0, lane);
#pragma unroll
    for (int j = 0; j < 4; ++j) {
      v16bf bj = load_frag_b(Bv + (size_t)(j * 16) * Sn, Sn, k0, lane);
      acc[0][j] = wmma_bf16(a0, bj, acc[0][j]);
      acc[1][j] = wmma_bf16(a1, bj, acc[1][j]);
    }
    acc[0][4] = wmma_bf16(a0, onesf, acc[0][4]);
    acc[1][4] = wmma_bf16(a1, onesf, acc[1][4]);
  }
#pragma unroll
  for (int i = 0; i < 2; ++i)
#pragma unroll
    for (int j = 0; j < 5; ++j)
#pragma unroll
      for (int r = 0; r < 8; ++r) {
        int m = m0 + i * 16 + r + ((lane >> 4) & 1) * 8;
        int n = j * 16 + (lane & 15);
        if (n <= 64)
          atomicAdd(&KVp[((size_t)bh * Mn + m) * DP + n], acc[i][j][r]);
      }
}

// KVTb[(bh*DP + n)*64 + m] = bf16(KVp[(bh*64 + m)*DP + n])
__global__ void k_kvt(const float* __restrict__ KVp, bf16* __restrict__ KVTb) {
  int idx = blockIdx.x * 256 + threadIdx.x;
  if (idx >= Bn * Hn * DP * Mn) return;
  int m = idx & 63;
  int n = (idx >> 6) % DP;
  int bh = idx / (DP * Mn);
  KVTb[idx] = (bf16)KVp[((size_t)bh * Mn + m) * DP + n];
}

// ---------------- 5) QKV + fused normalization ----------------
__global__ __launch_bounds__(256) void k_attn(
    const bf16* __restrict__ Qphi, const bf16* __restrict__ KVTb,
    bf16* __restrict__ attnb) {
  const int w = threadIdx.x >> 5, lane = threadIdx.x & 31;
  const int bh = blockIdx.z, b = bh >> 3, h = bh & 7;
  const int s0 = blockIdx.x * 256 + w * 32;

  const bf16* A  = Qphi + ((size_t)bh * Sn + s0) * Mn;   // lda = 64
  const bf16* BT = KVTb + (size_t)bh * DP * Mn;          // ldb = 64

  v8f acc[2][5] = {};
#pragma unroll
  for (int k0 = 0; k0 < Mn; k0 += 32) {
    v16bf a0 = load_frag_a(A, Mn, k0, lane);
    v16bf a1 = load_frag_a(A + (size_t)16 * Mn, Mn, k0, lane);
#pragma unroll
    for (int j = 0; j < 5; ++j) {
      v16bf bj = load_frag_b(BT + (size_t)(j * 16) * Mn, Mn, k0, lane);
      acc[0][j] = wmma_bf16(a0, bj, acc[0][j]);
      acc[1][j] = wmma_bf16(a1, bj, acc[1][j]);
    }
  }
#pragma unroll
  for (int i = 0; i < 2; ++i)
#pragma unroll
    for (int r = 0; r < 8; ++r) {
      // col 64 (QK_ones) sits at lane 0 (rows r) / lane 16 (rows r+8) of tile 4
      float div = __shfl(acc[i][4][r], lane & 16, 32);
      int s = s0 + i * 16 + r + ((lane >> 4) & 1) * 8;
#pragma unroll
      for (int j = 0; j < 4; ++j) {
        int d = j * 16 + (lane & 15);
        attnb[((size_t)(b * Sn + s)) * En + h * Dn + d] =
            (bf16)(acc[i][j][r] / div);
      }
    }
}

// ---------------- 6) output GEMM: out = attn @ Wo ----------------
__global__ __launch_bounds__(256) void k_out(
    const bf16* __restrict__ attnb, const bf16* __restrict__ WTo,
    float* __restrict__ out) {
  const int w = threadIdx.x >> 5, lane = threadIdx.x & 31;
  const int row0 = blockIdx.x * 128 + (w >> 1) * 32;
  const int col0 = blockIdx.y * 64 + (w & 1) * 32;

  const bf16* A = attnb + (size_t)row0 * En;
  v8f acc[2][2] = {};
#pragma unroll 4
  for (int k0 = 0; k0 < En; k0 += 32) {
    v16bf a0 = load_frag_a(A, En, k0, lane);
    v16bf a1 = load_frag_a(A + (size_t)16 * En, En, k0, lane);
    v16bf b0 = load_frag_b(WTo + (size_t)col0 * En, En, k0, lane);
    v16bf b1 = load_frag_b(WTo + (size_t)(col0 + 16) * En, En, k0, lane);
    acc[0][0] = wmma_bf16(a0, b0, acc[0][0]);
    acc[0][1] = wmma_bf16(a0, b1, acc[0][1]);
    acc[1][0] = wmma_bf16(a1, b0, acc[1][0]);
    acc[1][1] = wmma_bf16(a1, b1, acc[1][1]);
  }
#pragma unroll
  for (int i = 0; i < 2; ++i)
#pragma unroll
    for (int j = 0; j < 2; ++j)
#pragma unroll
      for (int r = 0; r < 8; ++r) {
        int row = row0 + i * 16 + r + ((lane >> 4) & 1) * 8;
        int col = col0 + j * 16 + (lane & 15);
        out[(size_t)row * En + col] = acc[i][j][r];
      }
}

// ---------------- host launcher ----------------
extern "C" void kernel_launch(void* const* d_in, const int* in_sizes, int n_in,
                              void* d_out, int out_size, void* d_ws, size_t ws_size,
                              hipStream_t stream) {
  const float* x   = (const float*)d_in[0];
  const float* Wq  = (const float*)d_in[1];
  const float* Wk  = (const float*)d_in[2];
  const float* Wv  = (const float*)d_in[3];
  const float* Wo  = (const float*)d_in[4];
  const float* omq = (const float*)d_in[5];
  const float* omk = (const float*)d_in[6];
  float* out = (float*)d_out;

  char* ws = (char*)d_ws;
  size_t off = 0;
  auto carve = [&](size_t bytes) -> char* {
    char* p = ws + off;
    off += (bytes + 255) & ~(size_t)255;
    return p;
  };
  bf16*  xb    = (bf16*)carve((size_t)NR * En * sizeof(bf16));
  bf16*  WTq   = (bf16*)carve((size_t)En * En * sizeof(bf16));
  bf16*  WTk   = (bf16*)carve((size_t)En * En * sizeof(bf16));
  bf16*  WTv   = (bf16*)carve((size_t)En * En * sizeof(bf16));
  bf16*  WTo   = (bf16*)carve((size_t)En * En * sizeof(bf16));
  bf16*  omqT  = (bf16*)carve((size_t)Dn * Mn * sizeof(bf16));
  bf16*  omkT  = (bf16*)carve((size_t)Dn * Mn * sizeof(bf16));
  bf16*  Qb    = (bf16*)carve((size_t)NR * En * sizeof(bf16));
  bf16*  Kb    = (bf16*)carve((size_t)NR * En * sizeof(bf16));
  bf16*  VT    = (bf16*)carve((size_t)NR * En * sizeof(bf16));
  float* qn2   = (float*)carve((size_t)NR * Hn * sizeof(float));
  float* kn2   = (float*)carve((size_t)NR * Hn * sizeof(float));
  bf16*  Qphi  = (bf16*)carve((size_t)Bn * Hn * Sn * Mn * sizeof(bf16));
  bf16*  KphiT = (bf16*)carve((size_t)Bn * Hn * Sn * Mn * sizeof(bf16));
  float* KVp   = (float*)carve((size_t)Bn * Hn * Mn * DP * sizeof(float));
  bf16*  KVTb  = (bf16*)carve((size_t)Bn * Hn * DP * Mn * sizeof(bf16));
  bf16*  attnb = xb;  // xb is dead after k_proj; reuse for attn

  // pack / transpose-convert
  {
    int n = NR * En;
    k_cvt_bf16<<<(n + 255) / 256, 256, 0, stream>>>(x, xb, n);
  }
  k_transpose_bf16<<<(En * En + 255) / 256, 256, 0, stream>>>(Wq, WTq, En, En);
  k_transpose_bf16<<<(En * En + 255) / 256, 256, 0, stream>>>(Wk, WTk, En, En);
  k_transpose_bf16<<<(En * En + 255) / 256, 256, 0, stream>>>(Wv, WTv, En, En);
  k_transpose_bf16<<<(En * En + 255) / 256, 256, 0, stream>>>(Wo, WTo, En, En);
  k_transpose_bf16<<<(Dn * Mn + 255) / 256, 256, 0, stream>>>(omq, omqT, Dn, Mn);
  k_transpose_bf16<<<(Dn * Mn + 255) / 256, 256, 0, stream>>>(omk, omkT, Dn, Mn);
  {
    int n = Bn * Hn * Mn * DP;
    k_zero_f32<<<(n + 255) / 256, 256, 0, stream>>>(KVp, n);
  }

  // Q/K/V projections (WMMA)
  k_proj<<<dim3(NR / 128, En / 64, 3), 256, 0, stream>>>(xb, WTq, WTk, WTv, Qb, Kb, VT);

  // row norms
  k_norm2<<<dim3((NR * Hn + 255) / 256, 1, 2), 256, 0, stream>>>(Qb, Kb, qn2, kn2);

  // feature maps (WMMA + exp epilogue)
  k_phi<<<dim3(Sn / 128, 1, 2 * Bn * Hn), 256, 0, stream>>>(
      Qb, Kb, omqT, omkT, qn2, kn2, Qphi, KphiT);

  // KV reduction (WMMA, split-S, f32 atomics)
  k_kv<<<dim3(Sn / 2048, 1, Bn * Hn), 256, 0, stream>>>(KphiT, VT, KVp);
  {
    int n = Bn * Hn * DP * Mn;
    k_kvt<<<(n + 255) / 256, 256, 0, stream>>>(KVp, KVTb);
  }

  // QKV + normalization (WMMA, shuffle broadcast of divisor)
  k_attn<<<dim3(Sn / 256, 1, Bn * Hn), 256, 0, stream>>>(Qphi, KVTb, attnb);

  // output projection (WMMA) -> f32 d_out
  k_out<<<dim3(NR / 128, En / 64, 1), 256, 0, stream>>>(attnb, WTo, out);
}
